// MultiHeadAttention_42614665511421
// MI455X (gfx1250) — matrix operand order repaired and checked by reading, not compiled, not run on hardware
//
#include <hip/hip_runtime.h>
#include <hip/hip_bf16.h>

// ---------------------------------------------------------------------------
// MHA on gfx1250 (MI455X): bf16 WMMA everywhere, f32 accumulation.
//   B=2, S=2048, D=1024, H=16, DK=64.
// Pipeline:
//   (0) cvt weights f32->bf16, mask -> additive bias (log2 domain)
//   (1) gemm_nt<f32A, bf16 out>     : qh = q @ Wq^T + bq      [B,S,D] bf16
//   (2) gemm_nt<f32A, bf16 out>     : kh = k @ Wk^T + bk      [B,S,D] bf16
//   (3) gemm_nt<f32A, bf16 transpT> : vhT = (v @ Wv^T + bv)^T [B,H,DK,S] bf16
//   (4) attn: flash-style online softmax in exp2 domain, 64-key chunks,
//       DPP ROW_XMASK reductions (VALU only), next-chunk global_prefetch,
//       WMMA scores + WMMA P.V
//   (5) gemm_nt<bf16A, f32 out>     : out = ctx @ Wo^T + bo   [B,S,D] f32
// ---------------------------------------------------------------------------

typedef __attribute__((ext_vector_type(16))) __bf16 v16bf;
typedef __attribute__((ext_vector_type(8)))  __bf16 v8bf;
typedef __attribute__((ext_vector_type(8)))  float  v8f;
typedef __attribute__((ext_vector_type(4)))  float  v4f;

#define B_DIM 2
#define S_DIM 2048
#define D_DIM 1024
#define H_DIM 16
#define DK_DIM 64
#define M_DIM (B_DIM * S_DIM)   // 4096

#define LOG2E 1.44269504088896340736f

__device__ __forceinline__ v8f wmma_bf16(v16bf a, v16bf b, v8f c) {
  return __builtin_amdgcn_wmma_f32_16x16x32_bf16(false, a, false, b, (short)0, c,
                                                 false, false);
}

// xor-lane reduction within 16-lane rows using DPP ROW_XMASK (0x160|mask):
// pure VALU crossbar, no LDS pipeline, no DScnt waits.
#define XMASK_MAX(v, m)                                                        \
  (v) = fmaxf((v), __int_as_float(__builtin_amdgcn_update_dpp(                 \
                  0, __float_as_int(v), 0x160 | (m), 0xf, 0xf, true)))
#define XMASK_ADD(v, m)                                                        \
  (v) = (v) + __int_as_float(__builtin_amdgcn_update_dpp(                      \
                  0, __float_as_int(v), 0x160 | (m), 0xf, 0xf, true))

// A-fragment (16x32, bf16, interleaved K layout):
//   lane<16 : elems 0..7 = K k0+0..7,  elems 8..15 = K k0+16..23
//   lane>=16: elems 0..7 = K k0+8..15, elems 8..15 = K k0+24..31
__device__ __forceinline__ v16bf load_a_frag_bf16(const __bf16* __restrict__ base,
                                                  int ld, int row0, int k0, int lane) {
  const __bf16* p = base + (size_t)(row0 + (lane & 15)) * ld + k0 + ((lane >> 4) << 3);
  v8bf lo = *(const v8bf*)(p);
  v8bf hi = *(const v8bf*)(p + 16);
  return __builtin_shufflevector(lo, hi, 0, 1, 2, 3, 4, 5, 6, 7,
                                 8, 9, 10, 11, 12, 13, 14, 15);
}

__device__ __forceinline__ v16bf load_a_frag_f32(const float* __restrict__ base,
                                                 int ld, int row0, int k0, int lane) {
  const float* p = base + (size_t)(row0 + (lane & 15)) * ld + k0 + ((lane >> 4) << 3);
  v4f x0 = *(const v4f*)(p + 0);
  v4f x1 = *(const v4f*)(p + 4);
  v4f y0 = *(const v4f*)(p + 16);
  v4f y1 = *(const v4f*)(p + 20);
  v16bf a;
#pragma unroll
  for (int i = 0; i < 4; ++i) {
    a[i]      = (__bf16)x0[i];
    a[4 + i]  = (__bf16)x1[i];
    a[8 + i]  = (__bf16)y0[i];
    a[12 + i] = (__bf16)y1[i];
  }
  return a;
}

// B-fragment (32x16, bf16). B stored NT (row n holds its K contiguously).
//   lane holds column n = col0 + (lane&15); lanes 0-15: K k0+0..7, k0+16..23,
//   lanes 16-31: K k0+8..15, k0+24..31 (interleaved, matches A) -> two 16B loads.
__device__ __forceinline__ v16bf load_b_frag(const __bf16* __restrict__ base,
                                             int ld, int col0, int k0, int lane) {
  const __bf16* p = base + (size_t)(col0 + (lane & 15)) * ld + k0 + ((lane >> 4) << 3);
  v8bf lo = *(const v8bf*)(p);
  v8bf hi = *(const v8bf*)(p + 16);
  return __builtin_shufflevector(lo, hi, 0, 1, 2, 3, 4, 5, 6, 7,
                                 8, 9, 10, 11, 12, 13, 14, 15);
}

// ---------------------------------------------------------------------------
// Converters
// ---------------------------------------------------------------------------
__global__ void cvt_f32_bf16_kernel(const float* __restrict__ src,
                                    __bf16* __restrict__ dst, int n) {
  int i = blockIdx.x * blockDim.x + threadIdx.x;
  if (i < n) dst[i] = (__bf16)src[i];
}

// Additive mask bias, pre-multiplied by log2(e) (softmax runs in exp2 domain).
__global__ void mask_bias_kernel(const int* __restrict__ mask,
                                 float* __restrict__ mb, int n) {
  int i = blockIdx.x * blockDim.x + threadIdx.x;
  if (i < n) mb[i] = (mask[i] == 0) ? (-1.0e9f * LOG2E) : 0.0f;
}

// ---------------------------------------------------------------------------
// NT GEMM: C[M,N] = A[M,K] * B[N,K]^T + bias[N]
//   AMODE : 0 = A is f32 (convert on load), 1 = A is bf16
//   OMODE : 0 = store bf16 row-major [M,N]
//           1 = store bf16 transposed-per-head: out[(b*H+h)*DK+d][s]  (for V)
//           2 = store f32 row-major [M,N]  (final output)
// Block: 256 threads (8 waves, 2x4), block tile 64x256, wave tile 32x64.
// ---------------------------------------------------------------------------
template <int AMODE, int OMODE>
__global__ void __launch_bounds__(256)
gemm_nt_kernel(const void* __restrict__ Av, const __bf16* __restrict__ Bw,
               const float* __restrict__ bias, void* __restrict__ Outv,
               int M, int N, int K) {
  const int lane = threadIdx.x & 31;
  const int wave = threadIdx.x >> 5;
  const int wm = wave & 1;   // 2 waves along M
  const int wn = wave >> 1;  // 4 waves along N
  const int m0 = blockIdx.x * 64 + wm * 32;
  const int n0 = blockIdx.y * 256 + wn * 64;

  v8f acc[2][4];
#pragma unroll
  for (int mt = 0; mt < 2; ++mt)
#pragma unroll
    for (int nt = 0; nt < 4; ++nt) acc[mt][nt] = (v8f){};

  const float*  Af = (const float*)Av;
  const __bf16* Ab = (const __bf16*)Av;

  for (int kk = 0; kk < K; kk += 32) {
    v16bf a[2], b[4];
#pragma unroll
    for (int mt = 0; mt < 2; ++mt) {
      if (AMODE == 0)
        a[mt] = load_a_frag_f32(Af, K, m0 + mt * 16, kk, lane);
      else
        a[mt] = load_a_frag_bf16(Ab, K, m0 + mt * 16, kk, lane);
    }
#pragma unroll
    for (int nt = 0; nt < 4; ++nt)
      b[nt] = load_b_frag(Bw, K, n0 + nt * 16, kk, lane);
#pragma unroll
    for (int mt = 0; mt < 2; ++mt)
#pragma unroll
      for (int nt = 0; nt < 4; ++nt)
        acc[mt][nt] = wmma_bf16(a[mt], b[nt], acc[mt][nt]);
  }

  // Epilogue. C layout: lane holds col n = (lane&15), rows r + 8*(lane>>4).
  const int half8 = (lane >> 4) << 3;
#pragma unroll
  for (int nt = 0; nt < 4; ++nt) {
    const int ncol = n0 + nt * 16 + (lane & 15);
    const float bv = bias[ncol];
#pragma unroll
    for (int mt = 0; mt < 2; ++mt) {
#pragma unroll
      for (int r = 0; r < 8; ++r) {
        const int m = m0 + mt * 16 + r + half8;
        const float c = acc[mt][nt][r] + bv;
        if (OMODE == 0) {
          ((__bf16*)Outv)[(size_t)m * N + ncol] = (__bf16)c;
        } else if (OMODE == 1) {
          const int bb = m / S_DIM, s = m - bb * S_DIM;
          const int h = ncol >> 6, d = ncol & 63;
          ((__bf16*)Outv)[((size_t)(bb * H_DIM + h) * DK_DIM + d) * S_DIM + s] =
              (__bf16)c;
        } else {
          ((float*)Outv)[(size_t)m * N + ncol] = c;
        }
      }
    }
  }
}

// ---------------------------------------------------------------------------
// Attention: one wave per (b, h, 16-row q tile). Online softmax over S in
// chunks of 64 keys: 8 score WMMAs + 8 P*V WMMAs per chunk; one softmax
// update per 64 keys. All exponentials in exp2 domain (bare v_exp_f32),
// row reductions via DPP ROW_XMASK, next chunk prefetched.
// ---------------------------------------------------------------------------
__global__ void __launch_bounds__(128)
attn_kernel(const __bf16* __restrict__ qh, const __bf16* __restrict__ kh,
            const __bf16* __restrict__ vhT, const float* __restrict__ mbias,
            __bf16* __restrict__ ctx) {
  const int lane = threadIdx.x & 31;
  const int wv = threadIdx.x >> 5;           // 4 waves/block
  const int wg = blockIdx.x * 4 + wv;        // 0 .. B*H*(S/16)-1 = 4095
  const int qt = wg & 127;                   // S/16 = 128 q tiles
  const int h  = (wg >> 7) & 15;
  const int b  = wg >> 11;
  const int q0 = qt * 16;

  const __bf16* qbase = qh + (size_t)b * S_DIM * D_DIM + h * DK_DIM;    // ld = D
  const __bf16* kbase = kh + (size_t)b * S_DIM * D_DIM + h * DK_DIM;    // ld = D
  const __bf16* vbase = vhT + (size_t)(b * H_DIM + h) * DK_DIM * S_DIM; // ld = S
  const float*  mb    = mbias + b * S_DIM;

  __shared__ __align__(16) __bf16 pbuf[4][16][64];  // per-wave P staging (2KB)

  // Q fragments are loop-invariant: 16 rows x 64 d -> two A frags.
  const v16bf qa0 = load_a_frag_bf16(qbase, D_DIM, q0, 0, lane);
  const v16bf qa1 = load_a_frag_bf16(qbase, D_DIM, q0, 32, lane);

  float mrow[8], lrow[8];
  v8f O[4];
#pragma unroll
  for (int r = 0; r < 8; ++r) { mrow[r] = -3.0e38f; lrow[r] = 0.0f; }
#pragma unroll
  for (int t = 0; t < 4; ++t) O[t] = (v8f){};

  const float scale2 = 0.125f * LOG2E;  // (1/sqrt(64)) * log2(e)
  const int half8 = (lane >> 4) << 3;

  for (int kc = 0; kc < S_DIM; kc += 64) {
    // ---- prefetch next chunk's K rows and V rows (global_prefetch_b8) ----
    if (kc + 64 < S_DIM) {
      const int nk = kc + 64;
      __builtin_prefetch(kbase + (size_t)(nk + lane) * D_DIM, 0, 3);
      __builtin_prefetch(kbase + (size_t)(nk + 32 + lane) * D_DIM, 0, 3);
      __builtin_prefetch(vbase + (size_t)lane * S_DIM + nk, 0, 3);
      __builtin_prefetch(vbase + (size_t)(32 + lane) * S_DIM + nk, 0, 3);
    }

    // ---- scores: S[16 q x 64 k] = Q (16x64) . K^T : 8 WMMAs ----
    v8f s[4];
#pragma unroll
    for (int ct = 0; ct < 4; ++ct) {
      v16bf ba = load_b_frag(kbase, D_DIM, kc + ct * 16, 0, lane);
      v16bf bb = load_b_frag(kbase, D_DIM, kc + ct * 16, 32, lane);
      s[ct] = wmma_bf16(qa1, bb, wmma_bf16(qa0, ba, (v8f){}));
    }

    float mbv[4];
#pragma unroll
    for (int ct = 0; ct < 4; ++ct) mbv[ct] = mb[kc + ct * 16 + (lane & 15)];

    // ---- online softmax update, exp2 domain ----
#pragma unroll
    for (int r = 0; r < 8; ++r) {
      float a0 = s[0][r] * scale2 + mbv[0];
      float a1 = s[1][r] * scale2 + mbv[1];
      float a2 = s[2][r] * scale2 + mbv[2];
      float a3 = s[3][r] * scale2 + mbv[3];
      float t = fmaxf(fmaxf(a0, a1), fmaxf(a2, a3));
      XMASK_MAX(t, 1);
      XMASK_MAX(t, 2);
      XMASK_MAX(t, 4);
      XMASK_MAX(t, 8);
      const float nm = fmaxf(mrow[r], t);
      const float corr = __builtin_amdgcn_exp2f(mrow[r] - nm);
      mrow[r] = nm;
      const float e0 = __builtin_amdgcn_exp2f(a0 - nm);
      const float e1 = __builtin_amdgcn_exp2f(a1 - nm);
      const float e2 = __builtin_amdgcn_exp2f(a2 - nm);
      const float e3 = __builtin_amdgcn_exp2f(a3 - nm);
      s[0][r] = e0;
      s[1][r] = e1;
      s[2][r] = e2;
      s[3][r] = e3;
      float rs = (e0 + e1) + (e2 + e3);
      XMASK_ADD(rs, 1);
      XMASK_ADD(rs, 2);
      XMASK_ADD(rs, 4);
      XMASK_ADD(rs, 8);
      lrow[r] = lrow[r] * corr + rs;
      O[0][r] *= corr;
      O[1][r] *= corr;
      O[2][r] *= corr;
      O[3][r] *= corr;
    }

    // ---- re-layout P: C-layout -> A-fragments, staged through LDS ----
#pragma unroll
    for (int r = 0; r < 8; ++r) {
#pragma unroll
      for (int ct = 0; ct < 4; ++ct)
        pbuf[wv][r + half8][ct * 16 + (lane & 15)] = (__bf16)s[ct][r];
    }
    asm volatile("s_wait_dscnt 0" ::: "memory");  // same-wave LDS RAW
    const __bf16* pp = &pbuf[wv][lane & 15][0];
    v8bf p0lo = *(const v8bf*)(pp + half8);
    v8bf p0hi = *(const v8bf*)(pp + 16 + half8);
    v8bf p1lo = *(const v8bf*)(pp + 32 + half8);
    v8bf p1hi = *(const v8bf*)(pp + 48 + half8);
    v16bf pa0 = __builtin_shufflevector(p0lo, p0hi, 0, 1, 2, 3, 4, 5, 6, 7,
                                        8, 9, 10, 11, 12, 13, 14, 15);
    v16bf pa1 = __builtin_shufflevector(p1lo, p1hi, 0, 1, 2, 3, 4, 5, 6, 7,
                                        8, 9, 10, 11, 12, 13, 14, 15);

    // ---- O += P (16x64) . V (64 x 64) : 8 WMMAs ----
#pragma unroll
    for (int t = 0; t < 4; ++t) {
      v16bf vb0 = load_b_frag(vbase, S_DIM, t * 16, kc, lane);
      v16bf vb1 = load_b_frag(vbase, S_DIM, t * 16, kc + 32, lane);
      O[t] = wmma_bf16(pa0, vb0, O[t]);
      O[t] = wmma_bf16(pa1, vb1, O[t]);
    }
  }

  // ---- normalize + store ctx[b, q, h*64+d] (bf16) ----
#pragma unroll
  for (int r = 0; r < 8; ++r) {
    const float inv = 1.0f / lrow[r];
    const int row = q0 + r + half8;
#pragma unroll
    for (int t = 0; t < 4; ++t) {
      const int d = t * 16 + (lane & 15);
      ctx[(size_t)(b * S_DIM + row) * D_DIM + h * DK_DIM + d] =
          (__bf16)(O[t][r] * inv);
    }
  }
}

// ---------------------------------------------------------------------------
// Host orchestration
// ---------------------------------------------------------------------------
extern "C" void kernel_launch(void* const* d_in, const int* in_sizes, int n_in,
                              void* d_out, int out_size, void* d_ws, size_t ws_size,
                              hipStream_t stream) {
  const float* q    = (const float*)d_in[0];
  const float* k    = (const float*)d_in[1];
  const float* v    = (const float*)d_in[2];
  const int*   mask = (const int*)d_in[3];
  const float* Wq   = (const float*)d_in[4];
  const float* bq   = (const float*)d_in[5];
  const float* Wk   = (const float*)d_in[6];
  const float* bk   = (const float*)d_in[7];
  const float* Wv   = (const float*)d_in[8];
  const float* bv   = (const float*)d_in[9];
  const float* Wo   = (const float*)d_in[10];
  const float* bo   = (const float*)d_in[11];
  float* out = (float*)d_out;

  // Workspace layout (bytes).
  char* ws = (char*)d_ws;
  const size_t szW   = (size_t)D_DIM * D_DIM * sizeof(__bf16);  // 2 MB
  const size_t szAct = (size_t)M_DIM * D_DIM * sizeof(__bf16);  // 8 MB
  size_t off = 0;
  __bf16* Wq_b  = (__bf16*)(ws + off); off += szW;
  __bf16* Wk_b  = (__bf16*)(ws + off); off += szW;
  __bf16* Wv_b  = (__bf16*)(ws + off); off += szW;
  __bf16* Wo_b  = (__bf16*)(ws + off); off += szW;
  __bf16* qh    = (__bf16*)(ws + off); off += szAct;
  __bf16* kh    = (__bf16*)(ws + off); off += szAct;
  __bf16* vhT   = (__bf16*)(ws + off); off += szAct;
  __bf16* ctx   = (__bf16*)(ws + off); off += szAct;
  float*  mbias = (float*)(ws + off);  off += (size_t)B_DIM * S_DIM * sizeof(float);
  (void)ws_size; (void)n_in; (void)in_sizes; (void)out_size;

  // (0) weight conversion + mask bias
  {
    const int nW = D_DIM * D_DIM;
    dim3 blk(256), grd((nW + 255) / 256);
    cvt_f32_bf16_kernel<<<grd, blk, 0, stream>>>(Wq, Wq_b, nW);
    cvt_f32_bf16_kernel<<<grd, blk, 0, stream>>>(Wk, Wk_b, nW);
    cvt_f32_bf16_kernel<<<grd, blk, 0, stream>>>(Wv, Wv_b, nW);
    cvt_f32_bf16_kernel<<<grd, blk, 0, stream>>>(Wo, Wo_b, nW);
    const int nM = B_DIM * S_DIM;
    mask_bias_kernel<<<(nM + 255) / 256, 256, 0, stream>>>(mask, mbias, nM);
  }

  // (1)-(3) QKV projections: grid 64 x 4, block 256
  {
    dim3 grd(M_DIM / 64, D_DIM / 256), blk(256);
    gemm_nt_kernel<0, 0><<<grd, blk, 0, stream>>>(q, Wq_b, bq, qh,
                                                  M_DIM, D_DIM, D_DIM);
    gemm_nt_kernel<0, 0><<<grd, blk, 0, stream>>>(k, Wk_b, bk, kh,
                                                  M_DIM, D_DIM, D_DIM);
    gemm_nt_kernel<0, 1><<<grd, blk, 0, stream>>>(v, Wv_b, bv, vhT,
                                                  M_DIM, D_DIM, D_DIM);
  }

  // (4) attention: B*H*(S/16) = 4096 waves, 4 waves/block
  attn_kernel<<<1024, 128, 0, stream>>>(qh, kh, vhT, mbias, ctx);

  // (5) output projection -> f32
  {
    dim3 grd(M_DIM / 64, D_DIM / 256), blk(256);
    gemm_nt_kernel<1, 2><<<grd, blk, 0, stream>>>(ctx, Wo_b, bo, out,
                                                  M_DIM, D_DIM, D_DIM);
  }
}
